// Net_7215545057326
// MI455X (gfx1250) — matrix operand order repaired and checked by reading, compiled, not run on hardware
//
#include <hip/hip_runtime.h>
#include <math.h>

typedef __attribute__((ext_vector_type(16))) _Float16 v16h;
typedef __attribute__((ext_vector_type(8)))  _Float16 v8h;
typedef __attribute__((ext_vector_type(8)))  float    v8f;

#define ELU(v) ((v) > 0.f ? (v) : (expf(v) - 1.f))

// ---------------------------------------------------------------- fill
__global__ void fill_f32(float* p, float v, int n) {
    int i = blockIdx.x * 256 + threadIdx.x;
    if (i < n) p[i] = v;
}

// ---------------------------------------------------------------- f32 -> f16 (straight)
__global__ void cvt_f16(const float* s, _Float16* d, int n) {
    int i = blockIdx.x * 256 + threadIdx.x;
    if (i < n) d[i] = (_Float16)s[i];
}

// f32 [K,N] -> f16 transposed [N,K]  (weight matrices for WMMA B operand)
__global__ void cvt_f16_T(const float* __restrict__ s, _Float16* __restrict__ d,
                          int Kd, int Nd, int n) {
    int i = blockIdx.x * 256 + threadIdx.x;
    if (i >= n) return;
    int k = i / Nd, c = i % Nd;
    d[(size_t)c * Kd + k] = (_Float16)s[i];
}

// fc2_w [128,10] -> padded transposed f16 [16,128]
__global__ void pad_fc2_f16T(const float* __restrict__ w, _Float16* __restrict__ d) {
    int i = blockIdx.x * 256 + threadIdx.x;
    if (i >= 16 * 128) return;
    int c = i >> 7, k = i & 127;
    d[i] = (c < 10) ? (_Float16)w[k * 10 + c] : (_Float16)0.f;
}

// ---------------------------------------------------------------- spline basis helper
__device__ __forceinline__ void spline_basis(float px, float py,
                                             float b[4], int w[4]) {
    px *= 4.f; py *= 4.f;
    float fx0 = floorf(px), fy0 = floorf(py);
    int ix0 = min(max((int)fx0, 0), 4), iy0 = min(max((int)fy0, 0), 4);
    int ix1 = min(ix0 + 1, 4),          iy1 = min(iy0 + 1, 4);
    float fx = px - fx0, fy = py - fy0;
    b[0] = (1.f - fx) * (1.f - fy);  w[0] = ix0 + 5 * iy0;
    b[1] = (1.f - fx) * fy;          w[1] = ix0 + 5 * iy1;
    b[2] = fx * (1.f - fy);          w[2] = ix1 + 5 * iy0;
    b[3] = fx * fy;                  w[3] = ix1 + 5 * iy1;
}

// ---------------------------------------------------------------- layer 1 edges (Cin=1, Cout=32)
__global__ void spline1_edge(const float* __restrict__ x,
                             const int* __restrict__ src, const int* __restrict__ dst,
                             const float* __restrict__ pseudo,
                             const float* __restrict__ W1,
                             float* agg, float* cnt, int E) {
    int t = blockIdx.x * 256 + threadIdx.x;
    int e = t >> 5, o = t & 31;
    if (e >= E) return;
    float b[4]; int w[4];
    spline_basis(pseudo[2 * e], pseudo[2 * e + 1], b, w);
    float ws = b[0] * W1[w[0] * 32 + o] + b[1] * W1[w[1] * 32 + o]
             + b[2] * W1[w[2] * 32 + o] + b[3] * W1[w[3] * 32 + o];
    float msg = x[src[e]] * ws;
    atomicAdd(&agg[(size_t)dst[e] * 32 + o], msg);
    if (o == 0) atomicAdd(&cnt[dst[e]], 1.f);
}

// mean + root + bias + ELU, then cluster max-pool into h2
__global__ void spline1_post(const float* __restrict__ x,
                             const float* __restrict__ agg, const float* __restrict__ cnt,
                             const float* __restrict__ root1, const float* __restrict__ b1,
                             const int* __restrict__ cluster,
                             float* h2max, int N) {
    int t = blockIdx.x * 256 + threadIdx.x;
    int n = t >> 5, o = t & 31;
    if (n >= N) return;
    float mean = agg[(size_t)n * 32 + o] / fmaxf(cnt[n], 1.f);
    float v = mean + x[n] * root1[o] + b1[o];
    v = ELU(v);
    atomicMax(&h2max[(size_t)cluster[n] * 32 + o], v);  // global_atomic_max_num_f32
}

// ---------------------------------------------------------------- layer 2 edges (32 -> 64)
__global__ void spline2_edge(const float* __restrict__ h2,
                             const int* __restrict__ src, const int* __restrict__ dst,
                             const float* __restrict__ pseudo,
                             const float* __restrict__ W2,   // [25,32,64], hot in L2 (200 KB)
                             float* agg, float* cnt, int E) {
    int t = blockIdx.x * 256 + threadIdx.x;
    int e = t >> 6, o = t & 63;
    if (e >= E) return;
    const float* hrow = h2 + (size_t)src[e] * 32;
    __builtin_prefetch(hrow, 0, 3);                  // global_prefetch_b8: cover gather latency
    float b[4]; int w[4];
    spline_basis(pseudo[2 * e], pseudo[2 * e + 1], b, w);
#pragma unroll
    for (int s = 0; s < 4; ++s)
        __builtin_prefetch(W2 + (size_t)w[s] * 2048 + o, 0, 3);
    // wide gather of the 32-float source row (8 x global_load_b128)
    float h[32];
    const float4* h4 = (const float4*)hrow;
#pragma unroll
    for (int q = 0; q < 8; ++q) {
        float4 v = h4[q];
        h[4 * q + 0] = v.x; h[4 * q + 1] = v.y; h[4 * q + 2] = v.z; h[4 * q + 3] = v.w;
    }
    float m = 0.f;
#pragma unroll
    for (int s = 0; s < 4; ++s) {
        const float* Wk = W2 + (size_t)w[s] * 2048 + o;
        float acc = 0.f;
#pragma unroll
        for (int i = 0; i < 32; ++i) acc = fmaf(h[i], Wk[i * 64], acc);
        m = fmaf(b[s], acc, m);
    }
    atomicAdd(&agg[(size_t)dst[e] * 64 + o], m);
    if (o == 0) atomicAdd(&cnt[dst[e]], 1.f);
}

// mean + (h2@root2, precomputed via WMMA) + bias + ELU, then cluster max-pool
__global__ void spline2_post(const float* __restrict__ agg, const float* __restrict__ cnt,
                             const float* __restrict__ rootterm, const float* __restrict__ b2,
                             const int* __restrict__ cluster,
                             float* h3max, int N) {
    int t = blockIdx.x * 256 + threadIdx.x;
    int n = t >> 6, o = t & 63;
    if (n >= N) return;
    float mean = agg[(size_t)n * 64 + o] / fmaxf(cnt[n], 1.f);
    float v = mean + rootterm[(size_t)n * 64 + o] + b2[o];
    v = ELU(v);
    atomicMax(&h3max[(size_t)cluster[n] * 64 + o], v);
}

// ---------------------------------------------------------------- generic WMMA GEMM
// C[M,N] = A[M,K] * B[K,N];  A row-major f16 [M,K], BT = B transposed f16 [N,K], C f32.
// M,N multiples of 16, K multiple of 32. One wave per 16x16 tile; one
// v_wmma_f32_16x16x32_f16 per 32-wide K step; fragments via 4 x global_load_b128.
__global__ void wmma_gemm_f16(const _Float16* __restrict__ A, const _Float16* __restrict__ BT,
                              float* __restrict__ C, int M, int N, int K, int tilesN) {
    int tile = blockIdx.x;
    int tm = tile / tilesN, tn = tile % tilesN;
    int lane = threadIdx.x;                 // 0..31, wave32
    int row = tm * 16 + (lane & 15);        // A-matrix: lane holds one M row
    int col = tn * 16 + (lane & 15);        // B-matrix: lane holds one N col
    int kbA = (lane < 16) ? 0 : 8;          // ISA 16-bit A layout (16x32)
    int kbB = (lane < 16) ? 0 : 8;          // same k order as A
    const _Float16* Ar = A  + (size_t)row * K;
    const _Float16* Bc = BT + (size_t)col * K;
    v8f acc = {};
    for (int k0 = 0; k0 < K; k0 += 32) {
        v8h a0 = *(const v8h*)(Ar + k0 + kbA);        // K = kbA+0..7
        v8h a1 = *(const v8h*)(Ar + k0 + kbA + 16);   // K = kbA+16..23
        v8h b0 = *(const v8h*)(Bc + k0 + kbB);        // K = kbB+0..7
        v8h b1 = *(const v8h*)(Bc + k0 + kbB + 16);   // K = kbB+16..23
        v16h a = __builtin_shufflevector(a0, a1, 0,1,2,3,4,5,6,7,8,9,10,11,12,13,14,15);
        v16h b = __builtin_shufflevector(b0, b1, 0,1,2,3,4,5,6,7,8,9,10,11,12,13,14,15);
        acc = __builtin_amdgcn_wmma_f32_16x16x32_f16(
            false, a, false, b, (short)0, acc, false, false);
    }
    // C layout: VGPR r -> M = r (lanes 0-15) or r+8 (lanes 16-31), N = lane&15
    int mbase = tm * 16 + ((lane < 16) ? 0 : 8);
#pragma unroll
    for (int r = 0; r < 8; ++r)
        C[(size_t)(mbase + r) * N + col] = acc[r];
}

// ---------------------------------------------------------------- batch mean pool
__global__ void pool_accum(const float* __restrict__ h3, const int* __restrict__ batch,
                           float* sums, float* cnt, int N) {
    int t = blockIdx.x * 256 + threadIdx.x;
    int n = t >> 6, o = t & 63;
    if (n >= N) return;
    int b = batch[n];
    atomicAdd(&sums[(size_t)b * 64 + o], h3[(size_t)n * 64 + o]);
    if (o == 0) atomicAdd(&cnt[b], 1.f);
}

__global__ void pool_mean_to_f16(const float* __restrict__ sums, const float* __restrict__ cnt,
                                 _Float16* g16, int n) {
    int i = blockIdx.x * 256 + threadIdx.x;
    if (i >= n) return;
    g16[i] = (_Float16)(sums[i] / fmaxf(cnt[i >> 6], 1.f));
}

// fc1 bias + ELU + cast to f16 for the next WMMA
__global__ void bias_elu_f16(const float* __restrict__ in, const float* __restrict__ bias,
                             _Float16* out16, int n) {
    int i = blockIdx.x * 256 + threadIdx.x;
    if (i >= n) return;
    float v = in[i] + bias[i & 127];
    out16[i] = (_Float16)ELU(v);
}

// fc2 bias + log_softmax over 10 classes (logits padded to 16 cols)
__global__ void head_logsoftmax(const float* __restrict__ logits, const float* __restrict__ b,
                                float* out, int rows) {
    int r = blockIdx.x * 256 + threadIdx.x;
    if (r >= rows) return;
    float z[10], mx = -1e30f;
#pragma unroll
    for (int c = 0; c < 10; ++c) { z[c] = logits[r * 16 + c] + b[c]; mx = fmaxf(mx, z[c]); }
    float s = 0.f;
#pragma unroll
    for (int c = 0; c < 10; ++c) s += expf(z[c] - mx);
    float l = logf(s);
#pragma unroll
    for (int c = 0; c < 10; ++c) out[r * 10 + c] = z[c] - mx - l;
}

// ----------------------------------------------------------------
extern "C" void kernel_launch(void* const* d_in, const int* in_sizes, int n_in,
                              void* d_out, int out_size, void* d_ws, size_t ws_size,
                              hipStream_t stream) {
    const float* x        = (const float*)d_in[0];
    const int*   ei1      = (const int*)  d_in[1];
    const float* ea1      = (const float*)d_in[2];
    const int*   cluster1 = (const int*)  d_in[3];
    const int*   ei2      = (const int*)  d_in[4];
    const float* ea2      = (const float*)d_in[5];
    const int*   cluster2 = (const int*)  d_in[6];
    const int*   batch3   = (const int*)  d_in[7];
    const float* W1       = (const float*)d_in[8];
    const float* root1    = (const float*)d_in[9];
    const float* b1       = (const float*)d_in[10];
    const float* W2       = (const float*)d_in[11];
    const float* root2    = (const float*)d_in[12];
    const float* b2       = (const float*)d_in[13];
    const float* fc1_w    = (const float*)d_in[14];
    const float* fc1_b    = (const float*)d_in[15];
    const float* fc2_w    = (const float*)d_in[16];
    const float* fc2_b    = (const float*)d_in[17];
    float* out = (float*)d_out;

    const int N1 = 80000, N2 = 40000, N3 = 20000, B = 512;
    const int E1 = in_sizes[1] / 2, E2 = in_sizes[4] / 2;

    // ---- carve workspace (~45 MB total)
    char* p = (char*)d_ws;
    auto alloc = [&](size_t bytes) -> void* {
        void* r = (void*)p; p += (bytes + 255) & ~(size_t)255; return r;
    };
    float*     agg1     = (float*)    alloc((size_t)N1 * 32 * 4);
    float*     cnt1     = (float*)    alloc((size_t)N1 * 4);
    float*     h2       = (float*)    alloc((size_t)N2 * 32 * 4);   // cluster-max result
    _Float16*  h2f16    = (_Float16*) alloc((size_t)N2 * 32 * 2);
    _Float16*  root2T   = (_Float16*) alloc(64 * 32 * 2);           // [N=64,K=32]
    float*     rootterm = (float*)    alloc((size_t)N2 * 64 * 4);
    float*     agg2     = (float*)    alloc((size_t)N2 * 64 * 4);
    float*     cnt2     = (float*)    alloc((size_t)N2 * 4);
    float*     h3       = (float*)    alloc((size_t)N3 * 64 * 4);
    float*     psum     = (float*)    alloc((size_t)B * 64 * 4);
    float*     pcnt     = (float*)    alloc((size_t)B * 4);
    _Float16*  g16      = (_Float16*) alloc((size_t)B * 64 * 2);
    _Float16*  fc1wT    = (_Float16*) alloc(128 * 64 * 2);          // [N=128,K=64]
    float*     fc1out   = (float*)    alloc((size_t)B * 128 * 4);
    _Float16*  fc1o16   = (_Float16*) alloc((size_t)B * 128 * 2);
    _Float16*  fc2wT    = (_Float16*) alloc(16 * 128 * 2);          // [N=16,K=128], cols 10..15 zero
    float*     logits   = (float*)    alloc((size_t)B * 16 * 4);

    auto blks = [](long n) { return (int)((n + 255) / 256); };

    // ---- re-init scratch every call (harness does not re-poison between replays)
    fill_f32<<<blks((long)N1 * 32), 256, 0, stream>>>(agg1, 0.f, N1 * 32);
    fill_f32<<<blks(N1), 256, 0, stream>>>(cnt1, 0.f, N1);
    fill_f32<<<blks((long)N2 * 32), 256, 0, stream>>>(h2, -1e30f, N2 * 32);
    fill_f32<<<blks((long)N2 * 64), 256, 0, stream>>>(agg2, 0.f, N2 * 64);
    fill_f32<<<blks(N2), 256, 0, stream>>>(cnt2, 0.f, N2);
    fill_f32<<<blks((long)N3 * 64), 256, 0, stream>>>(h3, -1e30f, N3 * 64);
    fill_f32<<<blks(B * 64), 256, 0, stream>>>(psum, 0.f, B * 64);
    fill_f32<<<blks(B), 256, 0, stream>>>(pcnt, 0.f, B);

    // ---- weight casts (transposed) for WMMA B operands
    cvt_f16_T<<<blks(32 * 64), 256, 0, stream>>>(root2, root2T, 32, 64, 32 * 64);
    cvt_f16_T<<<blks(64 * 128), 256, 0, stream>>>(fc1_w, fc1wT, 64, 128, 64 * 128);
    pad_fc2_f16T<<<blks(16 * 128), 256, 0, stream>>>(fc2_w, fc2wT);

    // ---- layer 1: edge scatter (atomic-bound), post (mean+root+ELU+cluster max)
    spline1_edge<<<blks((long)E1 * 32), 256, 0, stream>>>(
        x, ei1, ei1 + E1, ea1, W1, agg1, cnt1, E1);
    spline1_post<<<blks((long)N1 * 32), 256, 0, stream>>>(
        x, agg1, cnt1, root1, b1, cluster1, h2, N1);

    // ---- layer 2: root term via WMMA GEMM [40000,32]x[32,64]
    cvt_f16<<<blks((long)N2 * 32), 256, 0, stream>>>(h2, h2f16, N2 * 32);
    wmma_gemm_f16<<<(N2 / 16) * (64 / 16), 32, 0, stream>>>(
        h2f16, root2T, rootterm, N2, 64, 32, 64 / 16);

    spline2_edge<<<blks((long)E2 * 64), 256, 0, stream>>>(
        h2, ei2, ei2 + E2, ea2, W2, agg2, cnt2, E2);
    spline2_post<<<blks((long)N2 * 64), 256, 0, stream>>>(
        agg2, cnt2, rootterm, b2, cluster2, h3, N2);

    // ---- batch mean pool -> g (f16)
    pool_accum<<<blks((long)N3 * 64), 256, 0, stream>>>(h3, batch3, psum, pcnt, N3);
    pool_mean_to_f16<<<blks(B * 64), 256, 0, stream>>>(psum, pcnt, g16, B * 64);

    // ---- fc1: [512,64]x[64,128] WMMA, then bias+ELU; fc2: [512,128]x[128,16] WMMA
    wmma_gemm_f16<<<(B / 16) * (128 / 16), 32, 0, stream>>>(
        g16, fc1wT, fc1out, B, 128, 64, 128 / 16);
    bias_elu_f16<<<blks((long)B * 128), 256, 0, stream>>>(fc1out, fc1_b, fc1o16, B * 128);
    wmma_gemm_f16<<<(B / 16) * (16 / 16), 32, 0, stream>>>(
        fc1o16, fc2wT, logits, B, 16, 128, 1);

    // ---- log_softmax head
    head_logsoftmax<<<blks(B), 256, 0, stream>>>(logits, fc2_b, out, B);
}
